// HardgroupAttentionV2_16441134809374
// MI455X (gfx1250) — matrix operand order repaired and checked by reading, not compiled, not run on hardware
//
#include <hip/hip_runtime.h>

// ---------------------------------------------------------------------------
// HardgroupAttentionV2 for gfx1250 (MI455X).
// B=2, N=1600 (=40*40), C=384, h=12, d=32, G=48, K(top)=96.
// ---------------------------------------------------------------------------

#define NQ      1600
#define CIN     384
#define NHEAD   12
#define HDIM    32
#define GNUM    48
#define TOPK    96
#define BATCH   2
#define BH      (BATCH*NHEAD)          // 24
#define NROWS   (BATCH*NQ)             // 3200

typedef __attribute__((ext_vector_type(16))) __bf16 v16bf;
typedef __attribute__((ext_vector_type(8)))  __bf16 v8bf;
typedef __attribute__((ext_vector_type(8)))  float  v8f;

union U16 { v16bf v; v8bf h[2]; };

static __device__ inline v8f wmma_bf16(v16bf a, v16bf b, v8f c) {
    // D = A(16x32 bf16) * B(32x16 bf16) + C(16x16 f32)
    return __builtin_amdgcn_wmma_f32_16x16x32_bf16(
        false, a, false, b, (short)0, c, false, false);
}

static __device__ inline unsigned f2u_order(float f) {
    unsigned b = __float_as_uint(f);
    return (b & 0x80000000u) ? ~b : (b | 0x80000000u);
}

// ---------------------------------------------------------------------------
// Kernel 0: f32 -> bf16 conversions + zero group accumulators
// ---------------------------------------------------------------------------
__global__ __launch_bounds__(256) void k_convert_init(
    const float* __restrict__ x, const float* __restrict__ wqkv,
    const float* __restrict__ wproj,
    __bf16* __restrict__ xbf, __bf16* __restrict__ wqkvbf,
    __bf16* __restrict__ wprojbf,
    float* __restrict__ qsum, float* __restrict__ cnt)
{
    int i = blockIdx.x * 256 + threadIdx.x;
    int stride = gridDim.x * 256;
    for (int t = i; t < NROWS*CIN;        t += stride) xbf[t]     = (__bf16)x[t];
    for (int t = i; t < 3*NHEAD*HDIM*CIN; t += stride) wqkvbf[t]  = (__bf16)wqkv[t];
    for (int t = i; t < CIN*NHEAD*HDIM;   t += stride) wprojbf[t] = (__bf16)wproj[t];
    for (int t = i; t < BH*GNUM*HDIM;     t += stride) qsum[t] = 0.0f;
    for (int t = i; t < BH*GNUM;          t += stride) cnt[t]  = 0.0f;
}

// ---------------------------------------------------------------------------
// Kernel 1: QKV projection. y[n,r] = sum_c x[n,c]*wqkv[r,c].
// One wave per 16x16 output tile; WMMA bf16 over K=384 (12 steps).
// Scatter epilogue into q/k (f32 + bf16, [bh][n][d]) and v^T (bf16, [bh][d][n]).
// ---------------------------------------------------------------------------
__global__ __launch_bounds__(256) void k_gemm_qkv(
    const __bf16* __restrict__ xbf, const __bf16* __restrict__ wqkvbf,
    float* __restrict__ qf, float* __restrict__ kf,
    __bf16* __restrict__ qb, __bf16* __restrict__ kb,
    __bf16* __restrict__ vT)
{
    const int NT = 3*NHEAD*HDIM/16;                 // 72 col tiles
    int wid  = (blockIdx.x * blockDim.x + threadIdx.x) >> 5;
    int lane = threadIdx.x & 31, half = lane >> 4, lr = lane & 15;
    int tm = wid / NT, tn = wid % NT;               // 200 x 72 tiles
    int m0 = tm * 16, r0 = tn * 16;

    const __bf16* arow = xbf    + (size_t)(m0 + lr) * CIN;
    const __bf16* brow = wqkvbf + (size_t)(r0 + lr) * CIN;
    v8f acc = {};
    #pragma unroll
    for (int cb = 0; cb < CIN; cb += 32) {
        U16 a;
        a.h[0] = *(const v8bf*)(arow + cb + half*16);
        a.h[1] = *(const v8bf*)(arow + cb + half*16 + 8);
        v16bf b = *(const v16bf*)(brow + cb + half*16);
        acc = wmma_bf16(a.v, b, acc);
    }
    int r = r0 + lr;
    int s = r / (NHEAD*HDIM);                       // 0=q 1=k 2=v (uniform per wave)
    int head = (r >> 5) % NHEAD;
    int dd = r & 31;
    #pragma unroll
    for (int v = 0; v < 8; v++) {
        int mrow = m0 + v + half*8;
        int b_ = mrow / NQ, n = mrow % NQ;
        int bh = b_*NHEAD + head;
        float val = acc[v];
        size_t base = ((size_t)(bh*NQ + n))*HDIM + dd;
        if (s == 0)      { qf[base] = val; qb[base] = (__bf16)val; }
        else if (s == 1) { kf[base] = val; kb[base] = (__bf16)val; }
        else             { vT[((size_t)(bh*HDIM + dd))*NQ + n] = (__bf16)val; }
    }
}

// ---------------------------------------------------------------------------
// Kernel 2: query->group routing (argmax, first-index tie-break) + group sums
// ---------------------------------------------------------------------------
__global__ __launch_bounds__(256) void k_route(
    const float* __restrict__ qf, const float* __restrict__ wgp,
    int* __restrict__ gidx, float* __restrict__ qsum, float* __restrict__ cnt)
{
    __shared__ float gpl[GNUM*HDIM];
    int bh = blockIdx.x, head = bh % NHEAD;
    for (int i = threadIdx.x; i < GNUM*HDIM; i += 256)
        gpl[i] = wgp[head*GNUM*HDIM + i];
    __syncthreads();
    for (int n = threadIdx.x; n < NQ; n += 256) {
        const float* qr = qf + ((size_t)(bh*NQ + n))*HDIM;
        float qv[HDIM];
        #pragma unroll
        for (int d = 0; d < HDIM; d++) qv[d] = qr[d];
        float best = -__builtin_inff(); int bi = 0;
        for (int g = 0; g < GNUM; g++) {
            float sc = 0.f;
            #pragma unroll
            for (int d = 0; d < HDIM; d++) sc += qv[d] * gpl[g*HDIM + d];
            if (sc > best) { best = sc; bi = g; }
        }
        gidx[bh*NQ + n] = bi;
        float* qs = qsum + (size_t)(bh*GNUM + bi)*HDIM;
        #pragma unroll
        for (int d = 0; d < HDIM; d++) atomicAdd(&qs[d], qv[d]);
        atomicAdd(&cnt[bh*GNUM + bi], 1.0f);
    }
}

__global__ __launch_bounds__(256) void k_qmean(
    const float* __restrict__ qsum, const float* __restrict__ cnt,
    float* __restrict__ qmean)
{
    int i = blockIdx.x*256 + threadIdx.x;
    if (i < BH*GNUM*HDIM) qmean[i] = qsum[i] / fmaxf(cnt[i >> 5], 1e-8f);
}

// ---------------------------------------------------------------------------
// Kernel 3: per (bh,g): score all keys, find top-96 mask via bitwise binary
// search on order-preserving uint map; ties broken by lowest index.
// ---------------------------------------------------------------------------
__global__ __launch_bounds__(256) void k_topk_mask(
    const float* __restrict__ qmean, const float* __restrict__ kf,
    unsigned char* __restrict__ mask)
{
    __shared__ float qm[HDIM];
    __shared__ unsigned su[NQ];
    __shared__ int redc;
    int bhg = blockIdx.x;                    // 0..BH*GNUM-1
    int bh = bhg / GNUM;
    int tid = threadIdx.x;
    if (tid < HDIM) qm[tid] = qmean[(size_t)bhg*HDIM + tid];
    __syncthreads();
    for (int n = tid; n < NQ; n += 256) {
        const float* kr = kf + ((size_t)(bh*NQ + n))*HDIM;
        float sc = 0.f;
        #pragma unroll
        for (int d = 0; d < HDIM; d++) sc += qm[d]*kr[d];
        su[n] = f2u_order(sc);
    }
    __syncthreads();
    unsigned lo = 0u, hi = 0xFFFFFFFFu;
    for (int it = 0; it < 32; it++) {
        unsigned mid = (lo < hi) ? (lo + ((hi - lo) >> 1) + 1u) : lo;
        if (tid == 0) redc = 0;
        __syncthreads();
        int c = 0;
        for (int n = tid; n < NQ; n += 256) c += (su[n] >= mid);
        atomicAdd(&redc, c);
        __syncthreads();
        int tot = redc;
        __syncthreads();
        if (lo < hi) { if (tot >= TOPK) lo = mid; else hi = mid - 1u; }
    }
    unsigned T = lo;
    if (tid == 0) redc = 0;
    __syncthreads();
    { int c = 0; for (int n = tid; n < NQ; n += 256) c += (su[n] > T);
      atomicAdd(&redc, c); }
    __syncthreads();
    int rem = TOPK - redc;
    unsigned char* mrow = mask + (size_t)bhg*NQ;
    for (int n = tid; n < NQ; n += 256) {
        unsigned u = su[n];
        unsigned char mv = 0;
        if (u > T) mv = 1;
        else if (u == T) {
            int rank = 0;
            for (int j = 0; j < n; j++) rank += (su[j] == T);
            mv = (rank < rem) ? 1 : 0;
        }
        mrow[n] = mv;
    }
}

// ---------------------------------------------------------------------------
// Kernel 4: fused masked flash attention. One wave per 16-query tile.
// Scores: WMMA(Q, K^T). Online masked softmax (row reductions via shfl_xor
// within 16-lane halves, matching C-layout). P -> LDS -> A-fragment, then
// PV via two WMMAs against v^T. Output renormalized by masked denominator.
// ---------------------------------------------------------------------------
__global__ __launch_bounds__(256) void k_attn(
    const __bf16* __restrict__ qb, const __bf16* __restrict__ kb,
    const __bf16* __restrict__ vT, const int* __restrict__ gidx,
    const unsigned char* __restrict__ mask, __bf16* __restrict__ obf)
{
    __shared__ __bf16 Pl[8][16][32];
    const float NEGINF = -__builtin_inff();
    const float scale = 0.17677669529663687f;   // 1/sqrt(32)

    int wl   = threadIdx.x >> 5;
    int wid  = blockIdx.x*8 + wl;               // 0..2399
    int lane = threadIdx.x & 31, half = lane >> 4, lr = lane & 15;
    int bh = wid / (NQ/16), qt = wid % (NQ/16);
    int qbase = qt*16;
    int b = bh / NHEAD, head = bh % NHEAD;

    // resident Q A-fragment (16 queries x 32 d)
    const __bf16* qrow = qb + ((size_t)(bh*NQ + qbase + lr))*HDIM;
    U16 aq;
    aq.h[0] = *(const v8bf*)(qrow + half*16);
    aq.h[1] = *(const v8bf*)(qrow + half*16 + 8);

    const unsigned char* mp[8];
    #pragma unroll
    for (int v = 0; v < 8; v++) {
        int row = qbase + v + half*8;
        int g = gidx[bh*NQ + row];
        mp[v] = mask + ((size_t)(bh*GNUM + g))*NQ;
    }

    float rm[8], den[8];
    v8f accA = {}, accB = {};
    #pragma unroll
    for (int v = 0; v < 8; v++) { rm[v] = NEGINF; den[v] = 0.f; }

    for (int kbv = 0; kbv < NQ; kbv += 32) {
        // score tiles for keys [kbv, kbv+16) and [kbv+16, kbv+32)
        const __bf16* k0 = kb + ((size_t)(bh*NQ + kbv + lr))*HDIM + half*16;
        const __bf16* k1 = kb + ((size_t)(bh*NQ + kbv + 16 + lr))*HDIM + half*16;
        v16bf bf0 = *(const v16bf*)k0;
        v16bf bf1 = *(const v16bf*)k1;
        v8f z = {};
        v8f c0 = wmma_bf16(aq.v, bf0, z);
        v8f c1 = wmma_bf16(aq.v, bf1, z);

        #pragma unroll
        for (int v = 0; v < 8; v++) {
            int col0 = kbv + lr, col1 = kbv + 16 + lr;
            float s0 = mp[v][col0] ? c0[v]*scale : NEGINF;
            float s1 = mp[v][col1] ? c1[v]*scale : NEGINF;
            float tm = fmaxf(s0, s1);
            #pragma unroll
            for (int off = 1; off < 16; off <<= 1)
                tm = fmaxf(tm, __shfl_xor(tm, off, 16));
            float nm = fmaxf(rm[v], tm);
            float f  = (nm == NEGINF) ? 1.0f : __expf(rm[v] - nm);
            float p0 = (nm == NEGINF) ? 0.0f : __expf(s0 - nm);
            float p1 = (nm == NEGINF) ? 0.0f : __expf(s1 - nm);
            float ts = p0 + p1;
            #pragma unroll
            for (int off = 1; off < 16; off <<= 1)
                ts += __shfl_xor(ts, off, 16);
            den[v] = den[v]*f + ts;
            rm[v]  = nm;
            accA[v] *= f;
            accB[v] *= f;
            Pl[wl][v + half*8][lr]      = (__bf16)p0;
            Pl[wl][v + half*8][16 + lr] = (__bf16)p1;
        }
        asm volatile("s_wait_dscnt 0" ::: "memory");   // P visible across lanes (in-wave)

        U16 pa;
        pa.h[0] = *(const v8bf*)&Pl[wl][lr][half*16];
        pa.h[1] = *(const v8bf*)&Pl[wl][lr][half*16 + 8];

        const __bf16* vr0 = vT + ((size_t)(bh*HDIM + lr))*NQ      + kbv + half*16;
        const __bf16* vr1 = vT + ((size_t)(bh*HDIM + 16 + lr))*NQ + kbv + half*16;
        v16bf vf0 = *(const v16bf*)vr0;
        v16bf vf1 = *(const v16bf*)vr1;
        accA = wmma_bf16(pa.v, vf0, accA);
        accB = wmma_bf16(pa.v, vf1, accB);
    }

    #pragma unroll
    for (int v = 0; v < 8; v++) {
        int row = qbase + v + half*8;
        float dn = fmaxf(den[v], 1e-30f);
        size_t ob = ((size_t)(b*NQ + row))*(NHEAD*HDIM) + head*HDIM;
        obf[ob + lr]      = (__bf16)(accA[v]/dn);
        obf[ob + 16 + lr] = (__bf16)(accB[v]/dn);
    }
}

// ---------------------------------------------------------------------------
// Kernel 5: output projection out[n,c] = sum_j o[n,j]*wproj[c,j] (WMMA)
// ---------------------------------------------------------------------------
__global__ __launch_bounds__(256) void k_gemm_out(
    const __bf16* __restrict__ obf, const __bf16* __restrict__ wprojbf,
    float* __restrict__ out)
{
    const int NT = CIN/16;                          // 24 col tiles
    int wid  = (blockIdx.x * blockDim.x + threadIdx.x) >> 5;
    int lane = threadIdx.x & 31, half = lane >> 4, lr = lane & 15;
    int tm = wid / NT, tn = wid % NT;               // 200 x 24 tiles
    int m0 = tm*16, r0 = tn*16;
    const __bf16* arow = obf     + (size_t)(m0 + lr)*(NHEAD*HDIM);
    const __bf16* brow = wprojbf + (size_t)(r0 + lr)*(NHEAD*HDIM);
    v8f acc = {};
    #pragma unroll
    for (int cb = 0; cb < NHEAD*HDIM; cb += 32) {
        U16 a;
        a.h[0] = *(const v8bf*)(arow + cb + half*16);
        a.h[1] = *(const v8bf*)(arow + cb + half*16 + 8);
        v16bf bb = *(const v16bf*)(brow + cb + half*16);
        acc = wmma_bf16(a.v, bb, acc);
    }
    #pragma unroll
    for (int v = 0; v < 8; v++) {
        int mrow = m0 + v + half*8;
        out[(size_t)mrow*CIN + r0 + lr] = acc[v];
    }
}

// ---------------------------------------------------------------------------
extern "C" void kernel_launch(void* const* d_in, const int* in_sizes, int n_in,
                              void* d_out, int out_size, void* d_ws, size_t ws_size,
                              hipStream_t stream) {
    const float* x     = (const float*)d_in[0];
    const float* wqkv  = (const float*)d_in[1];
    const float* wgp   = (const float*)d_in[2];
    const float* wproj = (const float*)d_in[3];
    float* out = (float*)d_out;

    char* ws = (char*)d_ws;
    size_t off = 0;
    auto alloc = [&](size_t bytes) {
        char* p = ws + off;
        off = (off + bytes + 255) & ~(size_t)255;
        return p;
    };
    __bf16* xbf     = (__bf16*)alloc((size_t)NROWS*CIN*2);
    __bf16* wqkvbf  = (__bf16*)alloc((size_t)3*NHEAD*HDIM*CIN*2);
    __bf16* wprojbf = (__bf16*)alloc((size_t)CIN*NHEAD*HDIM*2);
    float*  qf      = (float* )alloc((size_t)BH*NQ*HDIM*4);
    float*  kf      = (float* )alloc((size_t)BH*NQ*HDIM*4);
    __bf16* qb      = (__bf16*)alloc((size_t)BH*NQ*HDIM*2);
    __bf16* kb      = (__bf16*)alloc((size_t)BH*NQ*HDIM*2);
    __bf16* vT      = (__bf16*)alloc((size_t)BH*NQ*HDIM*2);
    int*    gidx    = (int*   )alloc((size_t)BH*NQ*4);
    float*  qsum    = (float* )alloc((size_t)BH*GNUM*HDIM*4);
    float*  cnt     = (float* )alloc((size_t)BH*GNUM*4);
    float*  qmean   = (float* )alloc((size_t)BH*GNUM*HDIM*4);
    unsigned char* mask = (unsigned char*)alloc((size_t)BH*GNUM*NQ);
    __bf16* obf     = (__bf16*)alloc((size_t)NROWS*NHEAD*HDIM*2);
    (void)ws_size; (void)in_sizes; (void)n_in; (void)out_size;

    k_convert_init<<<1200, 256, 0, stream>>>(x, wqkv, wproj, xbf, wqkvbf, wprojbf, qsum, cnt);
    // 200*72 = 14400 tiles, 8 waves/block -> 1800 blocks
    k_gemm_qkv<<<1800, 256, 0, stream>>>(xbf, wqkvbf, qf, kf, qb, kb, vT);
    k_route<<<BH, 256, 0, stream>>>(qf, wgp, gidx, qsum, cnt);
    k_qmean<<<(BH*GNUM*HDIM + 255)/256, 256, 0, stream>>>(qsum, cnt, qmean);
    k_topk_mask<<<BH*GNUM, 256, 0, stream>>>(qmean, kf, mask);
    // 24*100 = 2400 waves -> 300 blocks
    k_attn<<<300, 256, 0, stream>>>(qb, kb, vT, gidx, mask, obf);
    // 200*24 = 4800 tiles -> 600 blocks
    k_gemm_out<<<600, 256, 0, stream>>>(obf, wprojbf, out);
}
